// CrossAttention_35278861370040
// MI455X (gfx1250) — matrix operand order repaired and checked by reading, not compiled, not run on hardware
//
#include <hip/hip_runtime.h>
#include <hip/hip_bf16.h>

// ---------------------------------------------------------------------------
// CrossAttention for MI455X (gfx1250, wave32, WMMA bf16 + double-buffered
// async LDS staging).  B=2, SQ=SKV=2048, DIM=1024, H=16, HD=64
// ---------------------------------------------------------------------------

typedef __attribute__((ext_vector_type(16))) __bf16 v16bf;
typedef __attribute__((ext_vector_type(8)))  float  v8f;
typedef __attribute__((ext_vector_type(4)))  int    v4i;

#if __has_builtin(__builtin_amdgcn_global_load_async_to_lds_b128) && \
    __has_builtin(__builtin_amdgcn_s_wait_asynccnt)
#define USE_ASYNC_LDS 1
#else
#define USE_ASYNC_LDS 0
#endif

__device__ inline unsigned short bfbits(float f) {
    // round-to-nearest-even f32 -> bf16
    unsigned u = __builtin_bit_cast(unsigned, f);
    unsigned r = u + 0x7FFFu + ((u >> 16) & 1u);
    return (unsigned short)(r >> 16);
}
__device__ inline __bf16 usbf(unsigned short u) {
    return __builtin_bit_cast(__bf16, u);
}
__device__ inline unsigned int packbf(float a, float b) {
    return (unsigned int)bfbits(a) | ((unsigned int)bfbits(b) << 16);
}
__device__ inline v8f wmma_bf16(v16bf a, v16bf b, v8f c) {
    // (neg_a, A, neg_b, B, c_mod, C, reuse_a, reuse_b)
    return __builtin_amdgcn_wmma_f32_16x16x32_bf16(false, a, false, b,
                                                   (short)0, c, false, false);
}

#if USE_ASYNC_LDS
__device__ inline void async_copy_b128(const void* gsrc, void* ldst) {
    __builtin_amdgcn_global_load_async_to_lds_b128(
        (__attribute__((address_space(1))) v4i*)gsrc,
        (__attribute__((address_space(3))) v4i*)ldst, 0, 0);
}
#endif

// ---------------------------------------------------------------------------
// GEMM:  Y[M,N] = X[M,K] @ W[K,N] + bias[N]
// One wave computes a 32(M) x 64(N) tile: 2 A-frags x 4 B-frags = 8 WMMA per
// K-step of 32 (each B fragment read feeds two WMMAs).
// ---------------------------------------------------------------------------
template <bool XBF, bool OUTF32>
__global__ __launch_bounds__(32) void gemm_bias(const void* __restrict__ Xv,
                                                const float* __restrict__ W,
                                                const float* __restrict__ bias,
                                                void* __restrict__ Yv,
                                                int M, int N, int K) {
    __shared__ alignas(16) unsigned int Alds32[32 * 32 / 2];   // 32x32 bf16
    __shared__ alignas(16) unsigned int Blds32[32 * 64 / 2];   // 32x64 bf16
    unsigned short* Alds = (unsigned short*)Alds32;
    unsigned short* Blds = (unsigned short*)Blds32;

    const int lane = threadIdx.x;
    const bool hi  = lane >= 16;
    const int ln   = lane & 15;
    const int m0   = blockIdx.x * 32;
    const int n0   = blockIdx.y * 64;

    const float* Xf          = (const float*)Xv;
    const unsigned int* Xu32 = (const unsigned int*)Xv;

    v8f acc[2][4] = {};

    for (int k0 = 0; k0 < K; k0 += 32) {
        // ---- stage A tile (32 rows x 32 cols) as packed bf16 pairs ----
#pragma unroll
        for (int i = 0; i < 16; ++i) {
            int p   = i * 32 + lane;          // 0..511 uint slots
            int row = p >> 4;                 // 0..31
            int cp  = p & 15;                 // column pair 0..15
            unsigned int v;
            if constexpr (XBF) {
                v = Xu32[((size_t)(m0 + row) * K + k0) / 2 + cp];
            } else {
                const float* xp = Xf + (size_t)(m0 + row) * K + k0 + 2 * cp;
                v = packbf(xp[0], xp[1]);
            }
            Alds32[p] = v;
        }
        // ---- stage B tile (32 rows x 64 cols) ----
#pragma unroll 4
        for (int r = 0; r < 32; ++r) {
            const float* wp = W + (size_t)(k0 + r) * N + n0 + 2 * lane;
            Blds32[r * 32 + lane] = packbf(wp[0], wp[1]);
        }
        __syncthreads();

        // ---- build both A fragments (contiguous k = e + 16*hi, same order as B) ----
        v16bf a[2];
#pragma unroll
        for (int ti = 0; ti < 2; ++ti)
#pragma unroll
            for (int e = 0; e < 16; ++e) {
                int kk = e + (hi ? 16 : 0);
                a[ti][e] = usbf(Alds[(ti * 16 + ln) * 32 + kk]);
            }
        // ---- 4 N-chunks; each B fragment reused by both A tiles ----
#pragma unroll
        for (int j = 0; j < 4; ++j) {
            v16bf bfr;
#pragma unroll
            for (int e = 0; e < 16; ++e) {
                int kk = e + (hi ? 16 : 0);
                bfr[e] = usbf(Blds[kk * 64 + j * 16 + ln]);
            }
#pragma unroll
            for (int ti = 0; ti < 2; ++ti)
                acc[ti][j] = wmma_bf16(a[ti], bfr, acc[ti][j]);
        }
        __syncthreads();
    }

    // ---- epilogue: bias add + store (C/D layout: row = r + hi*8, col = ln) --
#pragma unroll
    for (int ti = 0; ti < 2; ++ti)
#pragma unroll
        for (int j = 0; j < 4; ++j) {
            int col  = n0 + j * 16 + ln;
            float bj = bias[col];
#pragma unroll
            for (int r = 0; r < 8; ++r) {
                int row = m0 + ti * 16 + r + (hi ? 8 : 0);
                float v = acc[ti][j][r] + bj;
                if constexpr (OUTF32)
                    ((float*)Yv)[(size_t)row * N + col] = v;
                else
                    ((unsigned short*)Yv)[(size_t)row * N + col] = bfbits(v);
            }
        }
}

// ---------------------------------------------------------------------------
// Flash attention: one wave per (batch, head, 16-row q tile); KV blocks of 64.
// K/V tiles double-buffered in LDS via GLOBAL_LOAD_ASYNC_TO_LDS_B128:
// block i+1's 32 copies are issued before computing block i, then
// s_wait_asynccnt(32) drains only block i's copies (overlap DMA with WMMA).
// Q,K,V,O are bf16 [B*S, 1024] with head h at column offset h*64.
// ---------------------------------------------------------------------------
__global__ __launch_bounds__(32) void flash_attn(const unsigned short* __restrict__ Q,
                                                 const unsigned short* __restrict__ Kg,
                                                 const unsigned short* __restrict__ Vg,
                                                 unsigned short* __restrict__ O) {
    __shared__ alignas(16) unsigned int Klds32[2][64 * 32];   // 2x (64kv x 64d bf16)
    __shared__ alignas(16) unsigned int Vlds32[2][64 * 32];   // 2x (64kv x 64d bf16)
    __shared__ alignas(16) unsigned short Plds[16 * 64];      // 16q x 64kv bf16

    const int lane = threadIdx.x;
    const bool hi  = lane >= 16;
    const int ln   = lane & 15;
    const int q0   = blockIdx.x * 16;
    const int h    = blockIdx.y;
    const int b    = blockIdx.z;

#if USE_ASYNC_LDS
    // Each lane moves 16 B; 32 lanes cover 4 rows per issue; 32 issues/block.
    auto stage_async = [&](int kv0, int buf) {
        int rr = lane >> 3;               // 0..3
        int cb = (lane & 7) * 16;         // byte offset within 128 B row
#pragma unroll
        for (int r0 = 0; r0 < 64; r0 += 4) {
            size_t gb = ((size_t)(b * 2048 + kv0 + r0 + rr) * 1024 + h * 64) * 2 + cb;
            int lb = (r0 + rr) * 128 + cb;
            async_copy_b128((const char*)Kg + gb, (char*)Klds32[buf] + lb);
            async_copy_b128((const char*)Vg + gb, (char*)Vlds32[buf] + lb);
        }
    };
#endif

    // ---- load Q fragments (16 x 64 => 2 A-frags over d) ----
    const size_t qrow = (size_t)(b * 2048 + q0 + ln) * 1024 + h * 64;
    v16bf aQ[2];
#pragma unroll
    for (int c = 0; c < 2; ++c)
#pragma unroll
        for (int e = 0; e < 16; ++e) {
            int kk   = e + (hi ? 16 : 0);
            aQ[c][e] = usbf(Q[qrow + c * 32 + kk]);
        }

    v8f Oacc[4] = {};
    float mprev[8], lsum[8];
#pragma unroll
    for (int r = 0; r < 8; ++r) { mprev[r] = -1e30f; lsum[r] = 0.0f; }
    const float scale = 0.125f;   // 1/sqrt(64)

#if USE_ASYNC_LDS
    stage_async(0, 0);            // preload first KV block
#endif

    for (int kv0 = 0; kv0 < 2048; kv0 += 64) {
        const int buf = (kv0 >> 6) & 1;

#if USE_ASYNC_LDS
        if (kv0 + 64 < 2048) {
            stage_async(kv0 + 64, buf ^ 1);          // prefetch next block
            __builtin_amdgcn_s_wait_asynccnt(32);    // drain current block only
        } else {
            __builtin_amdgcn_s_wait_asynccnt(0);     // last block: full drain
        }
        __syncthreads();
#else
        {
            const unsigned int* K32 = (const unsigned int*)Kg;
            const unsigned int* V32 = (const unsigned int*)Vg;
#pragma unroll 4
            for (int r = 0; r < 64; ++r) {
                size_t base = (size_t)(b * 2048 + kv0 + r) * 512 + h * 32 + lane;
                Klds32[buf][r * 32 + lane] = K32[base];
                Vlds32[buf][r * 32 + lane] = V32[base];
            }
        }
        __syncthreads();
#endif
        const unsigned short* Klds = (const unsigned short*)Klds32[buf];
        const unsigned short* Vlds = (const unsigned short*)Vlds32[buf];

        // ---- S = Q @ K^T : four 16x16 tiles, reduce over d=64 ----
        v8f S[4];
#pragma unroll
        for (int t = 0; t < 4; ++t) {
            v8f s = {};
#pragma unroll
            for (int c = 0; c < 2; ++c) {
                v16bf bK;
#pragma unroll
                for (int e = 0; e < 16; ++e) {
                    int kk = e + (hi ? 16 : 0);       // d within chunk
                    bK[e]  = usbf(Klds[(t * 16 + ln) * 64 + c * 32 + kk]);
                }
                s = wmma_bf16(aQ[c], bK, s);
            }
            S[t] = s;
        }

        // ---- online softmax over the 64 new columns (f32) ----
#pragma unroll
        for (int r = 0; r < 8; ++r) {
            float s0 = S[0][r] * scale;
            float s1 = S[1][r] * scale;
            float s2 = S[2][r] * scale;
            float s3 = S[3][r] * scale;
            float mx = fmaxf(fmaxf(s0, s1), fmaxf(s2, s3));
#pragma unroll
            for (int off = 8; off >= 1; off >>= 1)
                mx = fmaxf(mx, __shfl_xor(mx, off, 32));
            float mnew = fmaxf(mprev[r], mx);
            float p0 = __expf(s0 - mnew);
            float p1 = __expf(s1 - mnew);
            float p2 = __expf(s2 - mnew);
            float p3 = __expf(s3 - mnew);
            float rs = (p0 + p1) + (p2 + p3);
#pragma unroll
            for (int off = 8; off >= 1; off >>= 1)
                rs += __shfl_xor(rs, off, 32);
            float corr = __expf(mprev[r] - mnew);
            lsum[r]    = lsum[r] * corr + rs;
            mprev[r]   = mnew;
#pragma unroll
            for (int j = 0; j < 4; ++j) Oacc[j][r] *= corr;
            int prow = r + (hi ? 8 : 0);
            Plds[prow * 64 + ln]      = bfbits(p0);
            Plds[prow * 64 + 16 + ln] = bfbits(p1);
            Plds[prow * 64 + 32 + ln] = bfbits(p2);
            Plds[prow * 64 + 48 + ln] = bfbits(p3);
        }
        __syncthreads();

        // ---- O += P(16x64) @ V(64x64) : 2 kv-chunks x 4 d-chunks ----
#pragma unroll
        for (int c = 0; c < 2; ++c) {
            v16bf aP;
#pragma unroll
            for (int e = 0; e < 16; ++e) {
                int kk = e + (hi ? 16 : 0);
                aP[e]  = usbf(Plds[ln * 64 + c * 32 + kk]);
            }
#pragma unroll
            for (int j = 0; j < 4; ++j) {
                v16bf bV;
#pragma unroll
                for (int e = 0; e < 16; ++e) {
                    int kk = e + (hi ? 16 : 0);      // kv within chunk
                    bV[e]  = usbf(Vlds[(c * 32 + kk) * 64 + j * 16 + ln]);
                }
                Oacc[j] = wmma_bf16(aP, bV, Oacc[j]);
            }
        }
        __syncthreads();   // all LDS reads done before buffer reuse / P rewrite
    }

    // ---- normalize and store (bf16) ----
#pragma unroll
    for (int j = 0; j < 4; ++j) {
#pragma unroll
        for (int r = 0; r < 8; ++r) {
            int row = q0 + r + (hi ? 8 : 0);
            float v = Oacc[j][r] / lsum[r];
            O[(size_t)(b * 2048 + row) * 1024 + h * 64 + j * 16 + ln] = bfbits(v);
        }
    }
}

// ---------------------------------------------------------------------------
extern "C" void kernel_launch(void* const* d_in, const int* in_sizes, int n_in,
                              void* d_out, int out_size, void* d_ws, size_t ws_size,
                              hipStream_t stream) {
    (void)in_sizes; (void)n_in; (void)out_size; (void)ws_size;
    const float* context = (const float*)d_in[0];   // [2,2048,1024]
    const float* hidden  = (const float*)d_in[1];   // [2,2048,1024]
    const float* wq = (const float*)d_in[2];
    const float* bq = (const float*)d_in[3];
    const float* wk = (const float*)d_in[4];
    const float* bk = (const float*)d_in[5];
    const float* wv = (const float*)d_in[6];
    const float* bv = (const float*)d_in[7];
    const float* wo = (const float*)d_in[8];
    const float* bo = (const float*)d_in[9];
    float* out = (float*)d_out;

    const size_t MT = 4096ull * 1024ull;            // tokens * dim
    unsigned short* Qbf = (unsigned short*)d_ws;    // 8 MB each
    unsigned short* Kbf = Qbf + MT;
    unsigned short* Vbf = Kbf + MT;
    unsigned short* Obf = Vbf + MT;

    dim3 gg(4096 / 32, 1024 / 64);                  // (128, 16) wave-tiles
    gemm_bias<false, false><<<gg, 32, 0, stream>>>(hidden,  wq, bq, Qbf, 4096, 1024, 1024);
    gemm_bias<false, false><<<gg, 32, 0, stream>>>(context, wk, bk, Kbf, 4096, 1024, 1024);
    gemm_bias<false, false><<<gg, 32, 0, stream>>>(context, wv, bv, Vbf, 4096, 1024, 1024);

    flash_attn<<<dim3(128, 16, 2), 32, 0, stream>>>(Qbf, Kbf, Vbf, Obf);

    gemm_bias<true, true><<<gg, 32, 0, stream>>>(Obf, wo, bo, out, 4096, 1024, 1024);
}
